// Decoder_32890859552887
// MI455X (gfx1250) — matrix operand order repaired and checked by reading, not compiled, not run on hardware
//
#include <hip/hip_runtime.h>
#include <cstdint>
#include <cstddef>

typedef __bf16 bf16;
typedef __bf16 v16bf __attribute__((ext_vector_type(16)));
typedef __bf16 v8bf  __attribute__((ext_vector_type(8)));
typedef float  v8f   __attribute__((ext_vector_type(8)));

#define WMMA_BF16(a, b, c) \
  __builtin_amdgcn_wmma_f32_16x16x32_bf16(false, (a), false, (b), (short)0, (c), false, false)

// ---------------------------------------------------------------------------
// CDNA5 async global->LDS copy (16B per lane), tracked by ASYNCcnt.
// LDS address operand = low 32 bits of the generic pointer (LDS aperture is
// entirely in addr[63:32] per ISA 10.2 aperture rules).
// ---------------------------------------------------------------------------
__device__ __forceinline__ void async_copy16(void* lds_ptr, const bf16* gptr) {
  uint32_t lds = (uint32_t)(uintptr_t)lds_ptr;
  uint64_t ga = (uint64_t)(uintptr_t)gptr;
  asm volatile("global_load_async_to_lds_b128 %0, %1, off"
               :: "v"(lds), "v"(ga)
               : "memory");
}
__device__ __forceinline__ void wait_async0() {
  asm volatile("s_wait_asynccnt 0x0" ::: "memory");
}

__device__ __forceinline__ v16bf cat16(v8bf lo, v8bf hi) {
  return __builtin_shufflevector(lo, hi, 0, 1, 2, 3, 4, 5, 6, 7, 8, 9, 10, 11, 12, 13, 14, 15);
}

// A fragment (16x32 bf16, MxK), ISA 7.12.2: lane m = lane&15 holds row m; kh = lane>>4.
// element e<8 -> k = kh*8+e ; e>=8 -> k = 16+kh*8+(e-8)  => two contiguous 16B runs.
__device__ __forceinline__ v16bf load_afrag(const bf16* base, int ldr, int lane) {
  int m = lane & 15, kh = lane >> 4;
  const bf16* p = base + (size_t)m * ldr + kh * 8;
  v8bf lo = *(const v8bf*)p;
  v8bf hi = *(const v8bf*)(p + 16);
  return cat16(lo, hi);
}

// B fragment (32x16 bf16, KxN) read from an [N][K]-contiguous buffer:
// lane col n = lane&15; element e<8 -> k = kh*8+e ; e>=8 -> k = 16+kh*8+(e-8)  => two contiguous 16B runs.
__device__ __forceinline__ v16bf load_bfrag(const bf16* base, int ldr, int lane) {
  int n = lane & 15, kh = lane >> 4;
  const bf16* p = base + (size_t)n * ldr + kh * 8;
  v8bf lo = *(const v8bf*)p;
  v8bf hi = *(const v8bf*)(p + 16);
  return cat16(lo, hi);
}

// reductions across the 16-lane half of a wave32 (xor masks 1..8 never cross bit 4)
__device__ __forceinline__ float rmax16(float x) {
#pragma unroll
  for (int m = 1; m < 16; m <<= 1) x = fmaxf(x, __shfl_xor(x, m, 32));
  return x;
}
__device__ __forceinline__ float rsum16(float x) {
#pragma unroll
  for (int m = 1; m < 16; m <<= 1) x += __shfl_xor(x, m, 32);
  return x;
}

__device__ __forceinline__ float gelu_tanh(float x) {
  const float c = 0.7978845608028654f;  // sqrt(2/pi)
  float u = c * (x + 0.044715f * x * x * x);
  return 0.5f * x * (1.0f + tanhf(u));
}

// ---------------------------------------------------------------------------
// f32 -> bf16 conversion (straight)
// ---------------------------------------------------------------------------
__global__ __launch_bounds__(256) void cvt_f32_bf16(const float* __restrict__ in,
                                                    bf16* __restrict__ out, int n) {
  int i = blockIdx.x * blockDim.x + threadIdx.x;
  int stride = gridDim.x * blockDim.x;
  for (; i < n; i += stride) out[i] = (bf16)in[i];
}

// ---------------------------------------------------------------------------
// f32 [K,N] -> bf16 [N,K] transpose-convert (weights), 32x32 LDS tiles.
// ---------------------------------------------------------------------------
__global__ __launch_bounds__(256) void cvt_f32_bf16_T(const float* __restrict__ in,
                                                      bf16* __restrict__ out, int K, int N) {
  __shared__ float tile[32][33];
  const int k0 = blockIdx.y * 32, n0 = blockIdx.x * 32;
  const int tx = threadIdx.x & 31, ty = threadIdx.x >> 5;  // 8 rows per pass
#pragma unroll
  for (int i = 0; i < 32; i += 8)
    tile[ty + i][tx] = in[(size_t)(k0 + ty + i) * N + n0 + tx];
  __syncthreads();
#pragma unroll
  for (int i = 0; i < 32; i += 8)
    out[(size_t)(n0 + ty + i) * K + k0 + tx] = (bf16)tile[tx][ty + i];
}

// ---------------------------------------------------------------------------
// GEMM: C[M,N] = act(A[M,K] * B[K,N] + bias [+ residual]).
// A bf16 row-major [M,K]; Bt bf16 PRE-TRANSPOSED [N,K]. f32 accumulate.
// Block = 256 threads = 8 waves; block tile 256x128, wave tile 32x128.
// Double-buffered async global->LDS staging, k-step 32.
// Optional outputs: outF (f32 [M,N]), outB (bf16 [M,N]), outBT (bf16 [N,M]).
// ---------------------------------------------------------------------------
__global__ __launch_bounds__(256) void gemm_wmma_bf16(
    const bf16* __restrict__ A, const bf16* __restrict__ Bt,
    const float* __restrict__ bias, const float* __restrict__ residual,
    float* __restrict__ outF, bf16* __restrict__ outB, bf16* __restrict__ outBT,
    int M, int N, int K, int doGelu) {
  __shared__ bf16 As[2][256][40];  // [row][k], 80B stride: conflict-free b128 frags
  __shared__ bf16 Bs[2][128][40];  // [col][k]

  const int t = threadIdx.x, lane = t & 31, wave = t >> 5;
  const int row0 = blockIdx.y * 256, col0 = blockIdx.x * 128;

  v8f acc[2][8];
#pragma unroll
  for (int h = 0; h < 2; ++h)
#pragma unroll
    for (int j = 0; j < 8; ++j) acc[h][j] = {};

  // staging decomposition: A tile = 256 rows x 4 chunks, B tile = 128 cols x 4 chunks
  auto stage = [&](int bsel, int kk) {
#pragma unroll
    for (int i = 0; i < 4; ++i) {  // 4 A chunks per thread
      int id = t + i * 256;
      int r = id >> 2, seg = id & 3;
      async_copy16(&As[bsel][r][seg * 8], A + (size_t)(row0 + r) * K + kk + seg * 8);
    }
#pragma unroll
    for (int i = 0; i < 2; ++i) {  // 2 B chunks per thread
      int id = t + i * 256;
      int c = id >> 2, seg = id & 3;
      async_copy16(&Bs[bsel][c][seg * 8], Bt + (size_t)(col0 + c) * K + kk + seg * 8);
    }
  };

  stage(0, 0);
  int buf = 0;
  for (int k0 = 0; k0 < K; k0 += 32) {
    wait_async0();
    __syncthreads();
    if (k0 + 32 < K) stage(buf ^ 1, k0 + 32);

    v16bf af0 = load_afrag(&As[buf][wave * 32][0], 40, lane);
    v16bf af1 = load_afrag(&As[buf][wave * 32 + 16][0], 40, lane);
#pragma unroll
    for (int j = 0; j < 8; ++j) {
      v16bf bfg = load_bfrag(&Bs[buf][j * 16][0], 40, lane);
      acc[0][j] = WMMA_BF16(af0, bfg, acc[0][j]);
      acc[1][j] = WMMA_BF16(af1, bfg, acc[1][j]);
    }
    buf ^= 1;
  }

  // epilogue: C-frag = col (lane&15), rows (lane>=16 ? 8 : 0)+v
  const int nl = lane & 15;
  const int rbase = (lane >> 4) * 8;
#pragma unroll
  for (int h = 0; h < 2; ++h) {
#pragma unroll
    for (int j = 0; j < 8; ++j) {
      const int col = col0 + j * 16 + nl;
      const float bs = bias ? bias[col] : 0.0f;
#pragma unroll
      for (int v = 0; v < 8; ++v) {
        const int row = row0 + wave * 32 + h * 16 + rbase + v;
        float x = acc[h][j][v] + bs;
        if (residual) x += residual[(size_t)row * N + col];
        if (doGelu) x = gelu_tanh(x);
        const size_t idx = (size_t)row * N + col;
        if (outF) outF[idx] = x;
        if (outB) outB[idx] = (bf16)x;
        if (outBT) outBT[(size_t)col * M + row] = (bf16)x;
      }
    }
  }
}

// ---------------------------------------------------------------------------
// Flash attention: grid = (Lq/128, B*H), block = 256 = 8 waves.
// Wave owns 16 q-rows. Q/K bf16 [B*L, 1024] (head h at col h*64);
// V bf16 TRANSPOSED [1024, B*L] (written by the V-projection GEMM).
// Double-buffered async K/V staging. S = Q K^T * 0.125 + mask, online
// softmax, O += P V. Writes bf16 ctx [B*L, 1024].
// ---------------------------------------------------------------------------
__global__ __launch_bounds__(256) void attn_wmma_bf16(
    const bf16* __restrict__ Qb, const bf16* __restrict__ Kb, const bf16* __restrict__ Vt_g,
    const float* __restrict__ mask, bf16* __restrict__ ctx, int Lq, int Lk, int Mtot) {
  const int D = 1024;
  __shared__ bf16 Klds[2][32][72];     // [key][d], 144B stride
  __shared__ bf16 Vt[2][64][40];       // [d][key]
  __shared__ bf16 Plds[8][16][40];     // per-wave P bounce: [qrow][key]

  const int t = threadIdx.x, lane = t & 31, wave = t >> 5;
  const int b = blockIdx.y >> 4, h = blockIdx.y & 15;
  const int q0 = blockIdx.x * 128 + wave * 16;
  const size_t qrow0 = (size_t)b * Lq + q0;

  // Q fragments (d = 0..31 and 32..63) held in registers for the whole key loop
  const v16bf qf0 = load_afrag(Qb + qrow0 * D + h * 64 + 0, D, lane);
  const v16bf qf1 = load_afrag(Qb + qrow0 * D + h * 64 + 32, D, lane);

  v8f o[4];
#pragma unroll
  for (int c = 0; c < 4; ++c) o[c] = {};
  float mrow[8], lrow[8];
#pragma unroll
  for (int v = 0; v < 8; ++v) { mrow[v] = -3.0e38f; lrow[v] = 0.0f; }

  const int nl = lane & 15;
  const int rbase = (lane >> 4) * 8;
  const int kKey = t >> 3, kSeg = t & 7;  // K tile: 32 keys x 8 chunks = 256
  const int vD = t >> 2, vSeg = t & 3;    // V tile: 64 d-rows x 4 chunks = 256

  auto stage = [&](int bsel, int kk) {
    async_copy16(&Klds[bsel][kKey][kSeg * 8],
                 Kb + ((size_t)b * Lk + kk + kKey) * D + h * 64 + kSeg * 8);
    async_copy16(&Vt[bsel][vD][vSeg * 8],
                 Vt_g + (size_t)(h * 64 + vD) * Mtot + (size_t)b * Lk + kk + vSeg * 8);
  };

  stage(0, 0);
  int buf = 0;
  for (int k0 = 0; k0 < Lk; k0 += 32) {
    wait_async0();
    __syncthreads();
    if (k0 + 32 < Lk) stage(buf ^ 1, k0 + 32);

    // scores: two 16x16 key tiles, each 2 WMMAs over d
    v8f s[2];
#pragma unroll
    for (int kh2 = 0; kh2 < 2; ++kh2) {
      v16bf kf0 = load_bfrag(&Klds[buf][kh2 * 16][0], 72, lane);
      v16bf kf1 = load_bfrag(&Klds[buf][kh2 * 16][32], 72, lane);
      v8f ss = {};
      ss = WMMA_BF16(qf0, kf0, ss);
      ss = WMMA_BF16(qf1, kf1, ss);
      s[kh2] = ss;
    }

    // scale + mask + online softmax update
    const float scale = 0.125f;  // 1/sqrt(64)
#pragma unroll
    for (int v = 0; v < 8; ++v) {
      const int qr = q0 + rbase + v;
      float s0 = s[0][v] * scale + mask[((size_t)b * Lq + qr) * Lk + k0 + nl];
      float s1 = s[1][v] * scale + mask[((size_t)b * Lq + qr) * Lk + k0 + 16 + nl];
      float mnew = fmaxf(mrow[v], rmax16(fmaxf(s0, s1)));
      float p0 = __expf(s0 - mnew);
      float p1 = __expf(s1 - mnew);
      float alpha = __expf(mrow[v] - mnew);
      lrow[v] = lrow[v] * alpha + rsum16(p0 + p1);
      mrow[v] = mnew;
#pragma unroll
      for (int c = 0; c < 4; ++c) o[c][v] *= alpha;
      Plds[wave][rbase + v][nl] = (bf16)p0;
      Plds[wave][rbase + v][16 + nl] = (bf16)p1;
    }
    __builtin_amdgcn_wave_barrier();  // order P store -> P load within the wave

    // O += P * V
    v16bf pf = load_afrag(&Plds[wave][0][0], 40, lane);
#pragma unroll
    for (int c = 0; c < 4; ++c) {
      v16bf vf = load_bfrag(&Vt[buf][c * 16][0], 40, lane);
      o[c] = WMMA_BF16(pf, vf, o[c]);
    }
    buf ^= 1;
  }

  // finalize: O /= l, write bf16 ctx in [B*L, D] layout
#pragma unroll
  for (int c = 0; c < 4; ++c) {
#pragma unroll
    for (int v = 0; v < 8; ++v) {
      float val = o[c][v] / lrow[v];
      ctx[((size_t)b * Lq + q0 + rbase + v) * D + h * 64 + c * 16 + nl] = (bf16)val;
    }
  }
}

// ---------------------------------------------------------------------------
// LayerNorm over D=1024 per row; writes f32 (+ optional bf16 re-quant).
// ---------------------------------------------------------------------------
__global__ __launch_bounds__(256) void ln_kernel(
    const float* __restrict__ in, const float* __restrict__ g, const float* __restrict__ bta,
    float* __restrict__ outF, bf16* __restrict__ outB) {
  const int D = 1024;
  const int row = blockIdx.x, t = threadIdx.x;
  const float* p = in + (size_t)row * D;
  float x[4];
  float s = 0.f, ss = 0.f;
#pragma unroll
  for (int i = 0; i < 4; ++i) {
    x[i] = p[t + i * 256];
    s += x[i];
    ss += x[i] * x[i];
  }
#pragma unroll
  for (int m = 1; m < 32; m <<= 1) {
    s += __shfl_xor(s, m, 32);
    ss += __shfl_xor(ss, m, 32);
  }
  __shared__ float shs[8], shss[8];
  const int lane = t & 31, wave = t >> 5;
  if (lane == 0) { shs[wave] = s; shss[wave] = ss; }
  __syncthreads();
  s = 0.f; ss = 0.f;
#pragma unroll
  for (int w = 0; w < 8; ++w) { s += shs[w]; ss += shss[w]; }
  const float mu = s * (1.0f / D);
  const float var = ss * (1.0f / D) - mu * mu;
  const float ri = rsqrtf(var + 1e-5f);
#pragma unroll
  for (int i = 0; i < 4; ++i) {
    const int idx = t + i * 256;
    const float y = (x[i] - mu) * ri * g[idx] + bta[idx];
    if (outF) outF[(size_t)row * D + idx] = y;
    if (outB) outB[(size_t)row * D + idx] = (bf16)y;
  }
}

// ---------------------------------------------------------------------------
// Host orchestration
// ---------------------------------------------------------------------------
extern "C" void kernel_launch(void* const* d_in, const int* in_sizes, int n_in,
                              void* d_out, int out_size, void* d_ws, size_t ws_size,
                              hipStream_t stream) {
  (void)in_sizes; (void)n_in; (void)out_size; (void)ws_size;
  const int Bsz = 2, L = 2048, D = 1024;
  const int M = Bsz * L;  // 4096 token rows

  const float* x_f32 = (const float*)d_in[0];
  const float* enc_f32 = (const float*)d_in[1];
  const float* self_mask = (const float*)d_in[2];
  const float* cross_mask = (const float*)d_in[3];
  const float* ff_w1 = (const float*)d_in[20];
  const float* ff_b1 = (const float*)d_in[21];
  const float* ff_w2 = (const float*)d_in[22];
  const float* ff_b2 = (const float*)d_in[23];

  char* w = (char*)d_ws;
  size_t off = 0;
  auto alloc = [&](size_t bytes) -> void* {
    void* p = w + off;
    off += (bytes + 255) & ~(size_t)255;
    return p;
  };

  bf16* wb[8];  // attention weights, TRANSPOSED [N][K] bf16
  for (int i = 0; i < 8; ++i) wb[i] = (bf16*)alloc((size_t)D * D * 2);
  bf16* ff1b = (bf16*)alloc((size_t)D * 4 * D * 2);  // [4D][D]
  bf16* ff2b = (bf16*)alloc((size_t)4 * D * D * 2);  // [D][4D]
  bf16* xb = (bf16*)alloc((size_t)M * D * 2);
  bf16* encb = (bf16*)alloc((size_t)M * D * 2);
  bf16* Qb = (bf16*)alloc((size_t)M * D * 2);
  bf16* Kb = (bf16*)alloc((size_t)M * D * 2);
  bf16* Vtb = (bf16*)alloc((size_t)M * D * 2);  // V transposed [D][M]
  bf16* ctx = (bf16*)alloc((size_t)M * D * 2);
  float* hbuf = (float*)alloc((size_t)M * D * 4);
  float* xres = (float*)alloc((size_t)M * D * 4);
  bf16* ffh = Qb;  // alias 32MB over dead Q/K/V/ctx during FFN

  auto cvt = [&](const float* src, bf16* dst, int n) {
    cvt_f32_bf16<<<1024, 256, 0, stream>>>(src, dst, n);
  };
  auto cvtT = [&](const float* src, bf16* dst, int K, int N) {
    dim3 grid(N / 32, K / 32);
    cvt_f32_bf16_T<<<grid, 256, 0, stream>>>(src, dst, K, N);
  };
  auto gemm = [&](const bf16* A, const bf16* BtW, const float* bias, const float* res,
                  float* oF, bf16* oB, bf16* oBT, int Mm, int Nn, int Kk, int act) {
    dim3 grid(Nn / 128, Mm / 256);
    gemm_wmma_bf16<<<grid, 256, 0, stream>>>(A, BtW, bias, res, oF, oB, oBT, Mm, Nn, Kk, act);
  };

  // weight conversion (transposed) + activation conversion
  for (int i = 0; i < 8; ++i) cvtT((const float*)d_in[4 + 2 * i], wb[i], D, D);
  cvtT(ff_w1, ff1b, D, 4 * D);
  cvtT(ff_w2, ff2b, 4 * D, D);
  cvt(x_f32, xb, M * D);
  cvt(enc_f32, encb, M * D);

  dim3 attnGrid(L / 128, Bsz * 16);

  // ---- self-attention block ----
  gemm(xb, wb[0], (const float*)d_in[5], nullptr, nullptr, Qb, nullptr, M, D, D, 0);
  gemm(xb, wb[1], (const float*)d_in[7], nullptr, nullptr, Kb, nullptr, M, D, D, 0);
  gemm(xb, wb[2], (const float*)d_in[9], nullptr, nullptr, nullptr, Vtb, M, D, D, 0);
  attn_wmma_bf16<<<attnGrid, 256, 0, stream>>>(Qb, Kb, Vtb, self_mask, ctx, L, L, M);
  gemm(ctx, wb[3], (const float*)d_in[11], x_f32, hbuf, nullptr, nullptr, M, D, D, 0);
  ln_kernel<<<M, 256, 0, stream>>>(hbuf, (const float*)d_in[24], (const float*)d_in[25],
                                   xres, xb);

  // ---- cross-attention block ----
  gemm(xb, wb[4], (const float*)d_in[13], nullptr, nullptr, Qb, nullptr, M, D, D, 0);
  gemm(encb, wb[5], (const float*)d_in[15], nullptr, nullptr, Kb, nullptr, M, D, D, 0);
  gemm(encb, wb[6], (const float*)d_in[17], nullptr, nullptr, nullptr, Vtb, M, D, D, 0);
  attn_wmma_bf16<<<attnGrid, 256, 0, stream>>>(Qb, Kb, Vtb, cross_mask, ctx, L, L, M);
  gemm(ctx, wb[7], (const float*)d_in[19], xres, hbuf, nullptr, nullptr, M, D, D, 0);
  ln_kernel<<<M, 256, 0, stream>>>(hbuf, (const float*)d_in[26], (const float*)d_in[27],
                                   xres, xb);

  // ---- feed-forward block ----
  gemm(xb, ff1b, ff_b1, nullptr, nullptr, ffh, nullptr, M, 4 * D, D, 1);  // gelu
  gemm(ffh, ff2b, ff_b2, xres, hbuf, nullptr, nullptr, M, D, 4 * D, 0);
  ln_kernel<<<M, 256, 0, stream>>>(hbuf, (const float*)d_in[28], (const float*)d_in[29],
                                   (float*)d_out, nullptr);
}
